// MultiHeadAttention_11106785427989
// MI455X (gfx1250) — matrix operand order repaired and checked by reading, not compiled, not run on hardware
//
#include <hip/hip_runtime.h>
#include <type_traits>

typedef __bf16 bf16;
typedef __attribute__((ext_vector_type(16))) __bf16 v16bf;
typedef __attribute__((ext_vector_type(8)))  __bf16 v8bf;
typedef __attribute__((ext_vector_type(4)))  __bf16 v4bf;
typedef __attribute__((ext_vector_type(8)))  float   v8f;
typedef __attribute__((ext_vector_type(4)))  float   v4f;

#define DEVINL __device__ __forceinline__

constexpr int HIDDEN   = 1024;
constexpr int HEADS    = 16;
constexpr int HEAD_DIM = 64;
constexpr int Bsz      = 2;
constexpr int SQn      = 2048;
constexpr int SKn      = 2048;
constexpr int Mrows    = Bsz * SQn;   // 4096 flattened rows

// ---------------------------------------------------------------------------
// Fragment loaders. Layouts per CDNA5 ISA 7.12.2 (wave32):
//  A (16x32, 16-bit): lane l<16 -> row l, elems {K0..7, K16..23};
//                     lane l>=16 -> row l-16, elems {K8..15, K24..31}
//  B (32x16, 16-bit): lane l -> col l%16; same K order as A
// ---------------------------------------------------------------------------
DEVINL v16bf load_afrag_f32(const float* p, int hl) {
  v4f x0 = *(const v4f*)(p + hl * 8);
  v4f x1 = *(const v4f*)(p + hl * 8 + 4);
  v4f x2 = *(const v4f*)(p + 16 + hl * 8);
  v4f x3 = *(const v4f*)(p + 16 + hl * 8 + 4);
  v16bf a;
#pragma unroll
  for (int i = 0; i < 4; ++i) {
    a[i]      = (bf16)x0[i];
    a[4 + i]  = (bf16)x1[i];
    a[8 + i]  = (bf16)x2[i];
    a[12 + i] = (bf16)x3[i];
  }
  return a;
}

DEVINL v16bf load_afrag_bf16(const bf16* __restrict__ p, int hl) {
  v8bf x0 = *(const v8bf*)(p + hl * 8);
  v8bf x1 = *(const v8bf*)(p + 16 + hl * 8);
  v16bf a;
#pragma unroll
  for (int i = 0; i < 8; ++i) { a[i] = x0[i]; a[8 + i] = x1[i]; }
  return a;
}

DEVINL v16bf load_bfrag_f32(const float* __restrict__ p, int hl) {
  v4f w0 = *(const v4f*)(p + hl * 8);
  v4f w1 = *(const v4f*)(p + hl * 8 + 4);
  v4f w2 = *(const v4f*)(p + 16 + hl * 8);
  v4f w3 = *(const v4f*)(p + 16 + hl * 8 + 4);
  v16bf b;
#pragma unroll
  for (int i = 0; i < 4; ++i) {
    b[i]      = (bf16)w0[i];
    b[4 + i]  = (bf16)w1[i];
    b[8 + i]  = (bf16)w2[i];
    b[12 + i] = (bf16)w3[i];
  }
  return b;
}

DEVINL v16bf load_bfrag_bf16(const bf16* __restrict__ p, int hl) {
  v8bf x0 = *(const v8bf*)(p + hl * 8);
  v8bf x1 = *(const v8bf*)(p + 16 + hl * 8);
  v16bf b;
#pragma unroll
  for (int i = 0; i < 8; ++i) { b[i] = x0[i]; b[8 + i] = x1[i]; }
  return b;
}

// ---------------------------------------------------------------------------
// Y[M,N] = (A[M,K] @ W[N,K]^T + bias) * out_scale   (K = N = HIDDEN).
// out_scale folds the attention softmax scale into the Q projection.
// 2x2 register blocking: one wave owns a 32x32 output tile (4 WMMA / k-step,
// each fragment feeds 2 WMMAs). Workgroup = 8 waves (2m x 4n) -> 64x128 tile.
// grid = (M/64, N/128), block = 256.
// TRANS_OUT: scatter-store V^T per (b,h): Y[((b*HEADS+h)*HEAD_DIM+d)*SKn + s]
// so the attention PV B-fragments become contiguous row loads.
// ---------------------------------------------------------------------------
template <typename AT, typename OT, bool TRANS_OUT>
__global__ __launch_bounds__(256) void gemm_xwT_bias(
    const AT* __restrict__ A, const float* __restrict__ W,
    const float* __restrict__ bias, OT* __restrict__ Y, float out_scale) {
  const int lane = threadIdx.x & 31;
  const int wave = threadIdx.x >> 5;
  const int hl   = lane >> 4;
  const int lm   = lane & 15;
  const int wm   = wave >> 2;      // 0..1
  const int wn   = wave & 3;       // 0..3
  const int mb   = blockIdx.x * 64 + wm * 32;
  const int nb   = blockIdx.y * 128 + wn * 32;

  const AT*    pA0 = A + (size_t)(mb + lm) * HIDDEN;
  const AT*    pA1 = A + (size_t)(mb + 16 + lm) * HIDDEN;
  const float* pW0 = W + (size_t)(nb + lm) * HIDDEN;
  const float* pW1 = W + (size_t)(nb + 16 + lm) * HIDDEN;

  v8f c00 = {}, c01 = {}, c10 = {}, c11 = {};
#pragma unroll 2
  for (int k0 = 0; k0 < HIDDEN; k0 += 32) {
    v16bf a0, a1;
    if constexpr (std::is_same_v<AT, float>) {
      a0 = load_afrag_f32(pA0 + k0, hl);
      a1 = load_afrag_f32(pA1 + k0, hl);
    } else {
      a0 = load_afrag_bf16(pA0 + k0, hl);
      a1 = load_afrag_bf16(pA1 + k0, hl);
    }
    const v16bf b0 = load_bfrag_f32(pW0 + k0, hl);
    const v16bf b1 = load_bfrag_f32(pW1 + k0, hl);
    c00 = __builtin_amdgcn_wmma_f32_16x16x32_bf16(false, a0, false, b0, (short)0, c00, false, false);
    c01 = __builtin_amdgcn_wmma_f32_16x16x32_bf16(false, a0, false, b1, (short)0, c01, false, false);
    c10 = __builtin_amdgcn_wmma_f32_16x16x32_bf16(false, a1, false, b0, (short)0, c10, false, false);
    c11 = __builtin_amdgcn_wmma_f32_16x16x32_bf16(false, a1, false, b1, (short)0, c11, false, false);
  }

  const v8f* cc[4] = { &c00, &c01, &c10, &c11 };
#pragma unroll
  for (int t = 0; t < 4; ++t) {
    const int mi = t >> 1, ni = t & 1;
    const int ncol = nb + ni * 16 + lm;
    const float bv = bias[ncol];
#pragma unroll
    for (int v = 0; v < 8; ++v) {  // C layout: VGPR v -> row v + hl*8, col lm
      const int row = mb + mi * 16 + v + hl * 8;
      const float val = ((*cc[t])[v] + bv) * out_scale;
      if constexpr (TRANS_OUT) {
        const int bidx = row >> 11, s = row & 2047;
        const int h = ncol >> 6, d = ncol & 63;
        Y[(((size_t)bidx * HEADS + h) * HEAD_DIM + d) * SKn + s] = (bf16)val;
      } else if constexpr (std::is_same_v<OT, float>) {
        Y[(size_t)row * HIDDEN + ncol] = val;
      } else {
        Y[(size_t)row * HIDDEN + ncol] = (bf16)val;
      }
    }
  }
}

// ---------------------------------------------------------------------------
// Attention: one workgroup (256 thr, 8 waves) = (batch b, 16 query rows),
// loops over all 16 heads. V is pre-transposed (Vt[b,h,d,k]); Q is pre-scaled
// by HEAD_DIM^-0.5 at projection time.
// LDS: P[16][2048] f32, mean[16][2048] f32, ctxp[8][16][64], scratch (~289KB,
// uses CDNA5's 320KB-per-WGP LDS).
// ---------------------------------------------------------------------------
__global__ __launch_bounds__(256) void attention_kernel(
    const bf16* __restrict__ Q, const bf16* __restrict__ Kp,
    const bf16* __restrict__ Vt, const unsigned char* __restrict__ mask,
    bf16* __restrict__ Cw, float* __restrict__ attn_mean) {
  extern __shared__ float smem[];
  float* P       = smem;                 // 16*2048
  float* meanacc = P + 16 * 2048;        // 16*2048
  float* ctxp    = meanacc + 16 * 2048;  // 8*16*64 = 8192
  float* red     = ctxp + 8192;          // 16*16
  float* rowstat = red + 256;            // 16

  const int tid  = threadIdx.x;
  const int lane = tid & 31;
  const int wave = tid >> 5;             // 0..7
  const int hl   = lane >> 4;
  const int lm   = lane & 15;
  const int b    = blockIdx.y;
  const int qb   = blockIdx.x * 16;
  const unsigned char* maskb = mask + b * SKn;

  // zero the mean accumulator (vectorized)
  for (int i = tid * 4; i < 16 * 2048; i += 256 * 4)
    *(v4f*)(meanacc + i) = (v4f){0.f, 0.f, 0.f, 0.f};

  // hoist per-lane mask bits (head-invariant): bit i <-> k-tile (i*8+wave)
  unsigned mbits = 0;
#pragma unroll 1
  for (int i = 0; i < 16; ++i) {
    const int kb = (i * 8 + wave) * 16;
    if (maskb[kb + lm] != 0) mbits |= (1u << i);
  }
  __syncthreads();

  for (int h = 0; h < HEADS; ++h) {
    // ---- Q fragments for this head (already scaled by HEAD_DIM^-0.5) ----
    const bf16* qptr = Q + ((size_t)(b * SQn + qb + lm)) * HIDDEN + h * HEAD_DIM;
    const v16bf qa0 = load_afrag_bf16(qptr,      hl);
    const v16bf qa1 = load_afrag_bf16(qptr + 32, hl);

    // ---- scores: 128 k-tiles split across 8 waves (literal trip count) ----
#pragma unroll 1
    for (int i = 0; i < 16; ++i) {
      const int kb = (i * 8 + wave) * 16;
      const bf16* kptr = Kp + ((size_t)(b * SKn + kb + lm)) * HIDDEN + h * HEAD_DIM;
      const v16bf kb0 = load_bfrag_bf16(kptr,      hl);
      const v16bf kb1 = load_bfrag_bf16(kptr + 32, hl);
      v8f c0 = {}, c1 = {};   // independent accumulators: no WMMA RAW chain
      c0 = __builtin_amdgcn_wmma_f32_16x16x32_bf16(false, qa0, false, kb0, (short)0, c0, false, false);
      c1 = __builtin_amdgcn_wmma_f32_16x16x32_bf16(false, qa1, false, kb1, (short)0, c1, false, false);
      const bool mk = (mbits >> i) & 1u;
#pragma unroll
      for (int v = 0; v < 8; ++v) {
        const int row = v + hl * 8;
        P[row * 2048 + kb + lm] = mk ? (c0[v] + c1[v]) : -1.0e30f;
      }
    }
    __syncthreads();

    // ---- row softmax: 16 threads per row, 128 cols each (vectorized) ----
    const int row = tid >> 4;
    const int seg = tid & 15;
    float* prow = P + row * 2048 + seg * 128;
    float* mrow = meanacc + row * 2048 + seg * 128;

    v4f mx4 = *(const v4f*)prow;
    for (int j = 4; j < 128; j += 4) {
      const v4f x = *(const v4f*)(prow + j);
#pragma unroll
      for (int e = 0; e < 4; ++e) mx4[e] = fmaxf(mx4[e], x[e]);
    }
    red[row * 16 + seg] = fmaxf(fmaxf(mx4[0], mx4[1]), fmaxf(mx4[2], mx4[3]));
    __syncthreads();
    if (seg == 0) {
      float m = red[row * 16];
      for (int j = 1; j < 16; ++j) m = fmaxf(m, red[row * 16 + j]);
      rowstat[row] = m;
    }
    __syncthreads();
    const float rm = rowstat[row];
    float s = 0.0f;
    for (int j = 0; j < 128; j += 4) {
      v4f x = *(const v4f*)(prow + j);
#pragma unroll
      for (int e = 0; e < 4; ++e) x[e] = __expf(x[e] - rm);
      *(v4f*)(prow + j) = x;
      s += x[0] + x[1] + x[2] + x[3];
    }
    red[row * 16 + seg] = s;
    __syncthreads();
    if (seg == 0) {
      float t = 0.0f;
      for (int j = 0; j < 16; ++j) t += red[row * 16 + j];
      rowstat[row] = (t > 0.0f) ? 1.0f / t : 0.0f;
    }
    __syncthreads();
    const float inv = rowstat[row];
    for (int j = 0; j < 128; j += 4) {
      v4f x = *(const v4f*)(prow + j);
      v4f m = *(const v4f*)(mrow + j);
#pragma unroll
      for (int e = 0; e < 4; ++e) {
        x[e] *= inv;
        m[e] += x[e] * (1.0f / (float)HEADS);
      }
      *(v4f*)(prow + j) = x;
      *(v4f*)(mrow + j) = m;
    }
    __syncthreads();

    // ---- ctx[16 x 64] = P @ V_h; 64 K-steps of 32 split across 8 waves ----
    v8f acc[4] = {};
#pragma unroll 1
    for (int i = 0; i < 8; ++i) {
      const int k0 = (i * 8 + wave) * 32;
      // A-fragment from LDS probs via b128 LDS loads + cvt
      const v16bf pa = load_afrag_f32(P + lm * 2048 + k0, hl);
      const bf16* vbase = Vt + (((size_t)b * HEADS + h) * HEAD_DIM) * SKn + k0;
#pragma unroll
      for (int dt = 0; dt < 4; ++dt) {
        // column d = dt*16+lm of B is a contiguous Vt row segment
        const v16bf vb = load_bfrag_bf16(vbase + (size_t)(dt * 16 + lm) * SKn, hl);
        acc[dt] = __builtin_amdgcn_wmma_f32_16x16x32_bf16(false, pa, false, vb,
                                                          (short)0, acc[dt], false, false);
      }
    }
#pragma unroll
    for (int dt = 0; dt < 4; ++dt)
#pragma unroll
      for (int v = 0; v < 8; ++v)
        ctxp[wave * 1024 + (v + hl * 8) * 64 + dt * 16 + lm] = acc[dt][v];
    __syncthreads();

    {  // cross-wave reduce + bf16 store, one v4f chunk per thread
      const int i = tid * 4;   // 1024 ctx elements / 256 threads
      v4f sum = {};
#pragma unroll
      for (int w = 0; w < 8; ++w) sum += *(const v4f*)(ctxp + w * 1024 + i);
      const int r = i >> 6, d = i & 63;
      v4bf o;
#pragma unroll
      for (int e = 0; e < 4; ++e) o[e] = (bf16)sum[e];
      *(v4bf*)(&Cw[((size_t)(b * SQn + qb + r)) * HIDDEN + h * HEAD_DIM + d]) = o;
    }
    __syncthreads();   // P / ctxp reused next head
  }

  // ---- write head-mean attention: attn_mean[b, q, k] (vectorized) ----
  for (int i = tid * 4; i < 16 * 2048; i += 256 * 4) {
    const v4f x = *(const v4f*)(meanacc + i);
    const int r = i >> 11, col = i & 2047;
    *(v4f*)(&attn_mean[((size_t)(b * SQn + qb + r)) * SKn + col]) = x;
  }
}

// ---------------------------------------------------------------------------
extern "C" void kernel_launch(void* const* d_in, const int* in_sizes, int n_in,
                              void* d_out, int out_size, void* d_ws, size_t ws_size,
                              hipStream_t stream) {
  const float* query = (const float*)d_in[0];
  const float* key_  = (const float*)d_in[1];
  const float* value = (const float*)d_in[2];
  const unsigned char* mask = (const unsigned char*)d_in[3];
  const float* Wq = (const float*)d_in[4];  const float* bq = (const float*)d_in[5];
  const float* Wk = (const float*)d_in[6];  const float* bk = (const float*)d_in[7];
  const float* Wv = (const float*)d_in[8];  const float* bv = (const float*)d_in[9];
  const float* Wo = (const float*)d_in[10]; const float* bo = (const float*)d_in[11];

  // workspace carve-out: Q, K, V^T, ctx in bf16 (4 x 8 MB = 32 MB)
  bf16* Qw  = (bf16*)d_ws;
  bf16* Kw  = Qw + (size_t)Mrows * HIDDEN;
  bf16* Vtw = Kw + (size_t)Mrows * HIDDEN;
  bf16* Cw  = Vtw + (size_t)Mrows * HIDDEN;

  float* out       = (float*)d_out;
  float* attn_mean = out + (size_t)Mrows * HIDDEN;

  const float qscale = 0.125f;   // HEAD_DIM^-0.5 folded into Q projection
  const dim3 gg(Mrows / 64, HIDDEN / 128);
  gemm_xwT_bias<float, bf16, false><<<gg, 256, 0, stream>>>(query, Wq, bq, Qw, qscale);
  gemm_xwT_bias<float, bf16, false><<<gg, 256, 0, stream>>>(key_,  Wk, bk, Kw, 1.0f);
  gemm_xwT_bias<float, bf16, true ><<<gg, 256, 0, stream>>>(value, Wv, bv, Vtw, 1.0f);

  const size_t smem =
      (size_t)(16 * 2048 + 16 * 2048 + 8192 + 256 + 16) * sizeof(float); // ~289 KB
  (void)hipFuncSetAttribute((const void*)attention_kernel,
                            hipFuncAttributeMaxDynamicSharedMemorySize, (int)smem);
  attention_kernel<<<dim3(SQn / 16, Bsz), 256, smem, stream>>>(
      Qw, Kw, Vtw, mask, Cw, attn_mean);

  gemm_xwT_bias<bf16, float, false><<<gg, 256, 0, stream>>>(Cw, Wo, bo, out, 1.0f);
}
